// HypergraphLayer_31275951849709
// MI455X (gfx1250) — matrix operand order repaired and checked by reading, not compiled, not run on hardware
//
#include <hip/hip_runtime.h>

// ---------------------------------------------------------------------------
// Types matching CDNA5 WMMA builtin signatures
// ---------------------------------------------------------------------------
typedef __bf16 bf16;
typedef __attribute__((ext_vector_type(16))) __bf16 v16bf;
typedef __attribute__((ext_vector_type(8)))  __bf16 v8bf;
typedef __attribute__((ext_vector_type(8)))  float   v8f;
typedef __attribute__((ext_vector_type(4)))  float   v4f;

#define N_NODES 20000
#define M_EDGES 4096
#define FDIM    128
#define NSEG    5   // K/N-loop segmentation for wave-level parallelism
// N_NODES/32 = 625 chunks -> 125 per segment; N_NODES/16 = 1250 tiles -> 250/seg

__device__ __forceinline__ v8f vzero8() {
  v8f z;
#pragma unroll
  for (int i = 0; i < 8; ++i) z[i] = 0.0f;
  return z;
}

__device__ __forceinline__ v8f wmma_bf16(v16bf a, v16bf b, v8f c) {
  // D = A(16x32 bf16) * B(32x16 bf16) + C(16x16 f32)
  return __builtin_amdgcn_wmma_f32_16x16x32_bf16(
      /*neg_a=*/false, a, /*neg_b=*/false, b,
      /*c_mod=*/(short)0, c, /*reuse_a=*/false, /*reuse_b=*/false);
}

// A-matrix (16x32, 16-bit) lane layout: lane l holds row M=l%16.
// element e: K = kbase + (e/8)*16 + (l/16)*8 + (e%8)
// -> two contiguous runs of 8 per lane. Source row-major fp32.
__device__ __forceinline__ v16bf load_a_f32(const float* __restrict__ row,
                                            int kbase, int half) {
  const v4f* p0 = (const v4f*)(row + kbase + half * 8);
  const v4f* p1 = (const v4f*)(row + kbase + 16 + half * 8);
  v4f x0 = p0[0], x1 = p0[1], x2 = p1[0], x3 = p1[1];
  v16bf a;
#pragma unroll
  for (int i = 0; i < 4; ++i) {
    a[i]      = (bf16)x0[i];
    a[i + 4]  = (bf16)x1[i];
    a[i + 8]  = (bf16)x2[i];
    a[i + 12] = (bf16)x3[i];
  }
  return a;
}

// Same A layout, source row-major bf16 (global or LDS).
__device__ __forceinline__ v16bf load_a_bf16(const bf16* row, int kbase, int half) {
  const v8bf* p0 = (const v8bf*)(row + kbase + half * 8);
  const v8bf* p1 = (const v8bf*)(row + kbase + 16 + half * 8);
  v8bf lo = *p0, hi = *p1;
  v16bf a;
#pragma unroll
  for (int i = 0; i < 8; ++i) {
    a[i]     = lo[i];
    a[i + 8] = hi[i];
  }
  return a;
}

// B-matrix (32x16, 16-bit) lane layout: lane l holds column N=l%16,
// element e: K = kbase + (e/8)*16 + (l/16)*8 + (e%8)  -> two runs of 8.
// colbase points at the K-contiguous column (base + N*ldK).
__device__ __forceinline__ v16bf load_b_bf16(const bf16* __restrict__ colbase,
                                             int kbase, int half) {
  const v8bf* p0 = (const v8bf*)(colbase + kbase + half * 8);
  const v8bf* p1 = (const v8bf*)(colbase + kbase + 16 + half * 8);
  v8bf lo = *p0, hi = *p1;
  v16bf b;
#pragma unroll
  for (int i = 0; i < 8; ++i) {
    b[i]     = lo[i];
    b[i + 8] = hi[i];
  }
  return b;
}

// ---------------------------------------------------------------------------
// K1: convert the four 128x128 weight matrices to transposed bf16
// ---------------------------------------------------------------------------
__global__ void k_convert_weights(const float* __restrict__ Wn,
                                  const float* __restrict__ We,
                                  const float* __restrict__ Wq,
                                  const float* __restrict__ Wk,
                                  bf16* __restrict__ WnT, bf16* __restrict__ WeT,
                                  bf16* __restrict__ WqT, bf16* __restrict__ WkT) {
  int idx = blockIdx.x * blockDim.x + threadIdx.x;
  if (idx < FDIM * FDIM) {
    int d = idx / FDIM, f = idx % FDIM;
    int t = f * FDIM + d;
    WnT[t] = (bf16)Wn[idx];
    WeT[t] = (bf16)We[idx];
    WqT[t] = (bf16)Wq[idx];
    WkT[t] = (bf16)Wk[idx];
  }
}

// ---------------------------------------------------------------------------
// K2: x_t = x @ W_node + b  (stored transposed bf16, K-contiguous for K3)
//     q   = x_t @ Wq + b    (stored row-major bf16 for K5/K6 A-loads)
// ---------------------------------------------------------------------------
__global__ void __launch_bounds__(32)
k_node_proj(const float* __restrict__ x,
            const bf16* __restrict__ WnT, const float* __restrict__ bn,
            const bf16* __restrict__ WqT, const float* __restrict__ bq,
            bf16* __restrict__ xtT, bf16* __restrict__ qout) {
  __shared__ __align__(16) bf16 lds_xt[16 * FDIM];
  const int l = threadIdx.x, half = l >> 4, lm = l & 15;
  const int n0 = blockIdx.x * 16;

  v8f acc[8];
#pragma unroll
  for (int ft = 0; ft < 8; ++ft) acc[ft] = vzero8();

  const float* xrow = x + (size_t)(n0 + lm) * FDIM;
#pragma unroll
  for (int dc = 0; dc < 4; ++dc) {
    v16bf a = load_a_f32(xrow, dc * 32, half);
#pragma unroll
    for (int ft = 0; ft < 8; ++ft) {
      v16bf b = load_b_bf16(WnT + (size_t)(ft * 16 + lm) * FDIM, dc * 32, half);
      acc[ft] = wmma_bf16(a, b, acc[ft]);
    }
  }

#pragma unroll
  for (int ft = 0; ft < 8; ++ft) {
    float bias = bn[ft * 16 + lm];
#pragma unroll
    for (int r = 0; r < 8; ++r) {
      int nloc = r + half * 8;
      bf16 v = (bf16)(acc[ft][r] + bias);
      xtT[(size_t)(ft * 16 + lm) * N_NODES + n0 + nloc] = v;   // K-contiguous
      lds_xt[nloc * FDIM + ft * 16 + lm] = v;                  // row-major tile
    }
  }
  __syncthreads();

  v8f qa[8];
#pragma unroll
  for (int ft = 0; ft < 8; ++ft) qa[ft] = vzero8();
  const bf16* ldsrow = lds_xt + lm * FDIM;  // A row n_local = lm
#pragma unroll
  for (int dc = 0; dc < 4; ++dc) {
    v16bf a = load_a_bf16(ldsrow, dc * 32, half);
#pragma unroll
    for (int ft = 0; ft < 8; ++ft) {
      v16bf b = load_b_bf16(WqT + (size_t)(ft * 16 + lm) * FDIM, dc * 32, half);
      qa[ft] = wmma_bf16(a, b, qa[ft]);
    }
  }
#pragma unroll
  for (int ft = 0; ft < 8; ++ft) {
    float bias = bq[ft * 16 + lm];
#pragma unroll
    for (int r = 0; r < 8; ++r)
      qout[(size_t)(n0 + r + half * 8) * FDIM + ft * 16 + lm] =
          (bf16)(qa[ft][r] + bias);
  }
}

// ---------------------------------------------------------------------------
// K3a: partial e_raw = H^T @ x_t over one K-segment (deterministic partials,
// no float atomics). grid = (M/16, NSEG); 1280 waves total.
// ---------------------------------------------------------------------------
__global__ void __launch_bounds__(32)
k_edge_agg_part(const float* __restrict__ H, const bf16* __restrict__ xtT,
                float* __restrict__ e_part) {
  const int l = threadIdx.x, half = l >> 4, lm = l & 15;
  const int m0 = blockIdx.x * 16;
  const int seg = blockIdx.y;
  const int m = m0 + lm;
  const int chunks = (N_NODES / 32) / NSEG;       // 125
  const int c0 = seg * chunks, c1 = c0 + chunks;

  v8f acc[8];
#pragma unroll
  for (int ft = 0; ft < 8; ++ft) acc[ft] = vzero8();

  for (int nc = c0; nc < c1; ++nc) {
    int n0 = nc * 32;
    if (nc + 2 < c1)  // stream hint for the big H matrix
      __builtin_prefetch(&H[(size_t)(n0 + 64 + half * 8) * M_EDGES + m], 0, 1);
    v16bf a;  // A = H^T: row m, K = n (strided scalar loads; H is binary)
#pragma unroll
    for (int i = 0; i < 8; ++i) {
      a[i]     = (bf16)H[(size_t)(n0 + half * 8 + i) * M_EDGES + m];
      a[i + 8] = (bf16)H[(size_t)(n0 + 16 + half * 8 + i) * M_EDGES + m];
    }
#pragma unroll
    for (int ft = 0; ft < 8; ++ft) {
      v16bf b = load_b_bf16(xtT + (size_t)(ft * 16 + lm) * N_NODES, n0, half);
      acc[ft] = wmma_bf16(a, b, acc[ft]);
    }
  }
  float* dst = e_part + (size_t)seg * M_EDGES * FDIM;
#pragma unroll
  for (int ft = 0; ft < 8; ++ft)
#pragma unroll
    for (int r = 0; r < 8; ++r)
      dst[(size_t)(m0 + r + half * 8) * FDIM + ft * 16 + lm] = acc[ft][r];
}

// K3b: e_raw = sum of NSEG partials (fixed order -> deterministic)
__global__ void k_edge_agg_reduce(const float* __restrict__ e_part,
                                  float* __restrict__ e_raw) {
  int idx = blockIdx.x * blockDim.x + threadIdx.x;
  if (idx < M_EDGES * FDIM) {
    float s = 0.0f;
#pragma unroll
    for (int seg = 0; seg < NSEG; ++seg)
      s += e_part[(size_t)seg * M_EDGES * FDIM + idx];
    e_raw[idx] = s;
  }
}

// ---------------------------------------------------------------------------
// K4: k = e_raw @ Wk + b   (row-major bf16; B of the logits GEMM)
//     e = relu(e_raw @ W_edge + b)  (stored transposed, K-contiguous for K6)
// ---------------------------------------------------------------------------
__global__ void __launch_bounds__(32)
k_edge_feat(const float* __restrict__ e_raw,
            const bf16* __restrict__ WkT, const float* __restrict__ bk,
            const bf16* __restrict__ WeT, const float* __restrict__ be,
            bf16* __restrict__ kout, bf16* __restrict__ eT) {
  const int l = threadIdx.x, half = l >> 4, lm = l & 15;
  const int m0 = blockIdx.x * 16;

  v8f ak[8], ae[8];
#pragma unroll
  for (int ft = 0; ft < 8; ++ft) { ak[ft] = vzero8(); ae[ft] = vzero8(); }

  const float* arow = e_raw + (size_t)(m0 + lm) * FDIM;
#pragma unroll
  for (int fc = 0; fc < 4; ++fc) {
    v16bf a = load_a_f32(arow, fc * 32, half);
#pragma unroll
    for (int ft = 0; ft < 8; ++ft) {
      ak[ft] = wmma_bf16(a, load_b_bf16(WkT + (size_t)(ft * 16 + lm) * FDIM, fc * 32, half), ak[ft]);
      ae[ft] = wmma_bf16(a, load_b_bf16(WeT + (size_t)(ft * 16 + lm) * FDIM, fc * 32, half), ae[ft]);
    }
  }
#pragma unroll
  for (int ft = 0; ft < 8; ++ft) {
    float bbk = bk[ft * 16 + lm], bbe = be[ft * 16 + lm];
#pragma unroll
    for (int r = 0; r < 8; ++r) {
      int mrow = m0 + r + half * 8;
      kout[(size_t)mrow * FDIM + ft * 16 + lm] = (bf16)(ak[ft][r] + bbk);
      float ev = ae[ft][r] + bbe;
      eT[(size_t)(ft * 16 + lm) * M_EDGES + mrow] = (bf16)(ev > 0.0f ? ev : 0.0f);
    }
  }
}

// ---------------------------------------------------------------------------
// K5a: ONLINE-softmax column stats over one N-segment. Single pass: running
// max + rescaled running sum (flash-attention style). Masked (H==0) entries
// skip the update, so empty columns produce sum 0 exactly like the reference.
// grid = (M/16, NSEG); logits recomputed via WMMA, never materialized.
// ---------------------------------------------------------------------------
__global__ void __launch_bounds__(32)
k_col_stats_part(const bf16* __restrict__ qb, const bf16* __restrict__ kb,
                 const float* __restrict__ H,
                 float* __restrict__ pmax, float* __restrict__ psum) {
  const int l = threadIdx.x, half = l >> 4, lm = l & 15;
  const int m0 = blockIdx.x * 16;
  const int seg = blockIdx.y;
  const float inv_scale = 1.0f / (sqrtf((float)FDIM) + 1e-7f);
  const int tiles = (N_NODES / 16) / NSEG;        // 250
  const int t0 = seg * tiles, t1 = t0 + tiles;

  v16bf bmat[4];  // k^T tiles: fixed per block, K=f contiguous in kb rows
#pragma unroll
  for (int fc = 0; fc < 4; ++fc)
    bmat[fc] = load_b_bf16(kb + (size_t)(m0 + lm) * FDIM, fc * 32, half);

  float runmax = -1e9f, runsum = 0.0f;
  for (int nt = t0; nt < t1; ++nt) {
    int n0 = nt * 16;
    v8f acc = vzero8();
#pragma unroll
    for (int fc = 0; fc < 4; ++fc)
      acc = wmma_bf16(load_a_bf16(qb + (size_t)(n0 + lm) * FDIM, fc * 32, half),
                      bmat[fc], acc);
#pragma unroll
    for (int r = 0; r < 8; ++r) {
      int n = n0 + r + half * 8;
      float h = H[(size_t)n * M_EDGES + m0 + lm];
      if (h > 0.0f) {                       // ~2% of entries
        float v = acc[r] * inv_scale;
        if (v > runmax) {
          runsum = runsum * __expf(runmax - v) + 1.0f;
          runmax = v;
        } else {
          runsum += __expf(v - runmax);
        }
      }
    }
  }
  // combine the two lane-halves holding the same column
  float om = __shfl_xor(runmax, 16, 32);
  float os = __shfl_xor(runsum, 16, 32);
  float gm = fmaxf(runmax, om);
  float gs = runsum * __expf(runmax - gm) + os * __expf(om - gm);
  if (l < 16) {
    pmax[(size_t)seg * M_EDGES + m0 + l] = gm;
    psum[(size_t)seg * M_EDGES + m0 + l] = gs;
  }
}

// K5b: merge segment partials: gmax = max_s, gsum = sum_s psum*exp(pmax-gmax)
__global__ void k_col_stats_reduce(const float* __restrict__ pmax,
                                   const float* __restrict__ psum,
                                   float* __restrict__ colmax,
                                   float* __restrict__ colsum) {
  int m = blockIdx.x * blockDim.x + threadIdx.x;
  if (m < M_EDGES) {
    float gm = -1e9f;
#pragma unroll
    for (int s = 0; s < NSEG; ++s)
      gm = fmaxf(gm, pmax[(size_t)s * M_EDGES + m]);
    float gs = 0.0f;
#pragma unroll
    for (int s = 0; s < NSEG; ++s)
      gs += psum[(size_t)s * M_EDGES + m] *
            __expf(pmax[(size_t)s * M_EDGES + m] - gm);
    colmax[m] = gm;
    colsum[m] = gs;
  }
}

// ---------------------------------------------------------------------------
// K6: out = attn @ e, fused LayerNorm. One wave owns a 16-row n-tile and all
// 128 output features. Streams m in chunks of 32: recompute logits (WMMA),
// build bf16 attn tile in LDS (C-layout -> A-layout conversion), then WMMA
// against eT. LN reduces over f with intra-wave shuffles.
// ---------------------------------------------------------------------------
__global__ void __launch_bounds__(32)
k_out_ln(const bf16* __restrict__ qb, const bf16* __restrict__ kb,
         const bf16* __restrict__ eT, const float* __restrict__ H,
         const float* __restrict__ colmax, const float* __restrict__ colsum,
         const float* __restrict__ gamma, const float* __restrict__ beta,
         float* __restrict__ out) {
  __shared__ __align__(16) bf16 attn_lds[16 * 32];
  const int l = threadIdx.x, half = l >> 4, lm = l & 15;
  const int n0 = blockIdx.x * 16;
  const float inv_scale = 1.0f / (sqrtf((float)FDIM) + 1e-7f);

  v16bf aq[4];  // q A-tiles, fixed per block
#pragma unroll
  for (int fc = 0; fc < 4; ++fc)
    aq[fc] = load_a_bf16(qb + (size_t)(n0 + lm) * FDIM, fc * 32, half);

  v8f oacc[8];
#pragma unroll
  for (int ft = 0; ft < 8; ++ft) oacc[ft] = vzero8();

  for (int mc = 0; mc < M_EDGES / 32; ++mc) {
#pragma unroll
    for (int sub = 0; sub < 2; ++sub) {
      int m0 = mc * 32 + sub * 16;
      v8f lacc = vzero8();
#pragma unroll
      for (int fc = 0; fc < 4; ++fc)
        lacc = wmma_bf16(aq[fc],
                         load_b_bf16(kb + (size_t)(m0 + lm) * FDIM, fc * 32, half),
                         lacc);
      float cmax = colmax[m0 + lm];
      float rsum = 1.0f / (colsum[m0 + lm] + 1e-7f);
#pragma unroll
      for (int r = 0; r < 8; ++r) {
        int n = n0 + r + half * 8;
        float h = H[(size_t)n * M_EDGES + m0 + lm];
        float a = (h > 0.0f) ? __expf(lacc[r] * inv_scale - cmax) * rsum : 0.0f;
        attn_lds[(r + half * 8) * 32 + sub * 16 + lm] = (bf16)a;
      }
    }
    __syncthreads();
    v16bf aat = load_a_bf16(attn_lds + lm * 32, 0, half);  // row n_local=lm
#pragma unroll
    for (int ft = 0; ft < 8; ++ft)
      oacc[ft] = wmma_bf16(aat,
                           load_b_bf16(eT + (size_t)(ft * 16 + lm) * M_EDGES,
                                       mc * 32, half),
                           oacc[ft]);
    __syncthreads();  // protect LDS before next chunk overwrites
  }

  // ---- fused LayerNorm over the 128 features of each row ----
#pragma unroll
  for (int r = 0; r < 8; ++r) {
    float s1 = 0.0f, s2 = 0.0f;
#pragma unroll
    for (int ft = 0; ft < 8; ++ft) {
      float v = oacc[ft][r];
      s1 += v;
      s2 += v * v;
    }
#pragma unroll
    for (int off = 1; off < 16; off <<= 1) {  // stays within 16-lane half
      s1 += __shfl_xor(s1, off, 32);
      s2 += __shfl_xor(s2, off, 32);
    }
    float mu = s1 * (1.0f / FDIM);
    float var = s2 * (1.0f / FDIM) - mu * mu;
    float rstd = rsqrtf(var + 1e-6f);
    int n = n0 + r + half * 8;
#pragma unroll
    for (int ft = 0; ft < 8; ++ft) {
      float g = gamma[ft * 16 + lm], b = beta[ft * 16 + lm];
      out[(size_t)n * FDIM + ft * 16 + lm] = (oacc[ft][r] - mu) * rstd * g + b;
    }
  }
}

// ---------------------------------------------------------------------------
// Host-side launcher
// ---------------------------------------------------------------------------
extern "C" void kernel_launch(void* const* d_in, const int* in_sizes, int n_in,
                              void* d_out, int out_size, void* d_ws, size_t ws_size,
                              hipStream_t stream) {
  const float* x       = (const float*)d_in[0];
  const float* H       = (const float*)d_in[1];
  const float* W_node  = (const float*)d_in[2];
  const float* b_node  = (const float*)d_in[3];
  const float* W_edge  = (const float*)d_in[4];
  const float* b_edge  = (const float*)d_in[5];
  const float* Wq      = (const float*)d_in[6];
  const float* bq      = (const float*)d_in[7];
  const float* Wk      = (const float*)d_in[8];
  const float* bk      = (const float*)d_in[9];
  const float* gamma   = (const float*)d_in[10];
  const float* beta    = (const float*)d_in[11];
  float* out = (float*)d_out;

  // workspace carve-up (all offsets 256B-aligned)
  char* p = (char*)d_ws;
  size_t off = 0;
  auto take = [&](size_t bytes) {
    char* r = p + off;
    off = (off + bytes + 255) & ~(size_t)255;
    return r;
  };
  bf16* WnT    = (bf16*)take((size_t)FDIM * FDIM * 2);
  bf16* WeT    = (bf16*)take((size_t)FDIM * FDIM * 2);
  bf16* WqT    = (bf16*)take((size_t)FDIM * FDIM * 2);
  bf16* WkT    = (bf16*)take((size_t)FDIM * FDIM * 2);
  bf16* xtT    = (bf16*)take((size_t)FDIM * N_NODES * 2);     // x_t transposed
  bf16* q_bf   = (bf16*)take((size_t)N_NODES * FDIM * 2);     // q row-major
  float* e_part = (float*)take((size_t)NSEG * M_EDGES * FDIM * 4);
  float* e_raw  = (float*)take((size_t)M_EDGES * FDIM * 4);
  bf16* k_bf   = (bf16*)take((size_t)M_EDGES * FDIM * 2);     // k row-major
  bf16* eT     = (bf16*)take((size_t)FDIM * M_EDGES * 2);     // e transposed
  float* pmax  = (float*)take((size_t)NSEG * M_EDGES * 4);
  float* psum  = (float*)take((size_t)NSEG * M_EDGES * 4);
  float* cmax  = (float*)take((size_t)M_EDGES * 4);
  float* csum  = (float*)take((size_t)M_EDGES * 4);
  (void)ws_size; (void)n_in; (void)in_sizes; (void)out_size;

  k_convert_weights<<<(FDIM * FDIM + 255) / 256, 256, 0, stream>>>(
      W_node, W_edge, Wq, Wk, WnT, WeT, WqT, WkT);

  k_node_proj<<<N_NODES / 16, 32, 0, stream>>>(x, WnT, b_node, WqT, bq, xtT, q_bf);

  k_edge_agg_part<<<dim3(M_EDGES / 16, NSEG), 32, 0, stream>>>(H, xtT, e_part);
  k_edge_agg_reduce<<<(M_EDGES * FDIM + 255) / 256, 256, 0, stream>>>(e_part, e_raw);

  k_edge_feat<<<M_EDGES / 16, 32, 0, stream>>>(e_raw, WkT, bk, WeT, b_edge,
                                               k_bf, eT);

  k_col_stats_part<<<dim3(M_EDGES / 16, NSEG), 32, 0, stream>>>(q_bf, k_bf, H,
                                                                pmax, psum);
  k_col_stats_reduce<<<(M_EDGES + 255) / 256, 256, 0, stream>>>(pmax, psum,
                                                                cmax, csum);

  k_out_ln<<<N_NODES / 16, 32, 0, stream>>>(q_bf, k_bf, eT, H, cmax, csum,
                                            gamma, beta, out);
}
